// TransMIL_56178172232086
// MI455X (gfx1250) — compile-verified, matrix operand order checked by reading, not run on hardware
//
#include <hip/hip_runtime.h>
#include <hip/hip_bf16.h>

// ---------------------------------------------------------------------------
// TransMIL forward for MI455X (gfx1250).
//  * Heavy GEMMs -> v_wmma_f32_16x16x32_bf16 (f32 accumulate), LDS-tiled
//    128x(128|64)x64, per-wave 64x32 register tile = 8 WMMA per K-chunk.
//  * A-tile staging via Tensor Data Mover (tensor_load_to_lds) with pad_enable
//    reproducing the bank-conflict-padded LDS stride (66) in hardware.
//  * Newton-Schulz pseudo-inverse kept fp32 -> v_wmma_f32_16x16x4_f32.
// Working set (<= 84MB per tensor) fits the 192MB L2 -> compute bound, so the
// win is WMMA occupancy, not bandwidth. Needs ~280 MB workspace (d_ws).
// ---------------------------------------------------------------------------

#define NP     10000   // patches
#define NTOK   10001   // +cls
#define NPAD   10240   // nystrom padded length (multiple of 256)
#define PADR   239     // front zero-pad rows
#define CDIM   512
#define NHEAD  8
#define DHEAD  64
#define NLM    256     // landmarks
#define LFOLD  40      // NPAD / NLM
#define LDSW   66      // LDS row stride (64 + 2 pad) in bf16 elements

typedef __attribute__((ext_vector_type(16))) __bf16 v16bf;
typedef __attribute__((ext_vector_type(8)))  float  v8f;
typedef __attribute__((ext_vector_type(2)))  float  v2f;
typedef __attribute__((ext_vector_type(4)))  unsigned int v4u;
typedef __attribute__((ext_vector_type(4)))  int    v4i;
typedef __attribute__((ext_vector_type(8)))  int    v8i;

#if defined(__has_builtin)
#  if __has_builtin(__builtin_amdgcn_wmma_f32_16x16x4_f32)
#    define HAVE_WMMA_F32X4 1
#  endif
#  if __has_builtin(__builtin_amdgcn_tensor_load_to_lds) && \
      __has_builtin(__builtin_amdgcn_s_wait_tensorcnt)
#    define HAVE_TDM 1
#  endif
#endif

__device__ __forceinline__ unsigned short f2bf(float f){
  unsigned int u = __float_as_uint(f);
  u += 0x7fffu + ((u >> 16) & 1u);
  return (unsigned short)(u >> 16);
}
__device__ __forceinline__ float bf2f(unsigned short s){
  return __uint_as_float(((unsigned int)s) << 16);
}

union FragB { v16bf v; unsigned int u[8]; };

// Load a 16-element bf16 WMMA operand from an LDS row (pairs are contiguous K).
// ISA 16-bit A layout: lane = row, half = lane>>4; element pair 2j,2j+1 is
// K = half*8 + 2j (+16 for j>=4).
__device__ __forceinline__ void frag_load(unsigned int* u, const unsigned short* row, int half){
  const unsigned short* p = row + half * 8;
  u[0] = *(const unsigned int*)(p + 0);
  u[1] = *(const unsigned int*)(p + 2);
  u[2] = *(const unsigned int*)(p + 4);
  u[3] = *(const unsigned int*)(p + 6);
  const unsigned short* q = row + 16 + half * 8;
  u[4] = *(const unsigned int*)(q + 0);
  u[5] = *(const unsigned int*)(q + 2);
  u[6] = *(const unsigned int*)(q + 4);
  u[7] = *(const unsigned int*)(q + 6);
}

// ---------------------------------------------------------------------------
// bf16 GEMM: C(MxN) = A(MxK) * B, f32 accumulate.
//   Tile: BM=128, BN=32*WN, BK=64. Waves: 2 (M) x WN (N); per-wave 64x32.
//   BT=true : B stored (N x K) row-major ; BT=false: B stored (K x N).
//   EPI: 0 = store f32*alpha; 1 = QKV split; 2 = bias + residual add.
// Requires M%128==0, N%BN==0, K%64==0 (true for every call here).
// ---------------------------------------------------------------------------
struct GemmP {
  const unsigned short* A; long lda, sA;
  const unsigned short* B; long ldb, sB;
  float* C; long ldc, sC;
  int M, N, K;
  const float* bias;
  float* vout; unsigned short* qout; unsigned short* kout; unsigned short* vbfout;
  float* Hres; int rowoff; long ldh;
  float alpha;
};

template<int EPI, bool BT, int WN>
__global__ __launch_bounds__(256) void gemm_bf16_k(GemmP p){
  constexpr int BN = 32 * WN;       // 128 or 64
  constexpr int NT = 64 * WN;       // threads: 256 or 128
  __shared__ unsigned short As[128 * LDSW];
  __shared__ unsigned short Bs[BN * LDSW];
  const int  tid   = threadIdx.x;
  const int  wv    = tid >> 5;
  const int  lane  = tid & 31;
  const int  waveM = wv & 1;        // 2 waves along M (64 rows each)
  const int  waveN = wv >> 1;       // WN waves along N (32 cols each)
  const int  half  = lane >> 4;
  const int  rr    = lane & 15;
  const long blockRow = (long)blockIdx.y * 128;
  const long blockCol = (long)blockIdx.x * BN;
  const unsigned short* A = p.A + (long)blockIdx.z * p.sA;
  const unsigned short* B = p.B + (long)blockIdx.z * p.sB;

  v8f zero = {};
  v8f acc[4][2];
  #pragma unroll
  for (int i = 0; i < 4; ++i){ acc[i][0] = zero; acc[i][1] = zero; }

  for (int k0 = 0; k0 < p.K; k0 += 64){
    // ---- stage A tile (128 x 64) ----
#if defined(HAVE_TDM)
    // Tensor Data Mover: 2D tile DMA global->LDS; pad_enable inserts 1 DWORD
    // after every 32 DWORDs (one 64-elt row) -> LDS row stride 66. Wave 0
    // issues; DMA overlaps the vector-staged B tile below.
    if (tid < 32){
      const unsigned long long ga =
          (unsigned long long)(const void*)(A + blockRow * p.lda + k0);
      const unsigned lds_off = (unsigned)(unsigned long long)(const void*)&As[0];
      v4u g0;
      g0.x = 1u;                                           // count=1 valid D#
      g0.y = lds_off;                                      // lds_addr
      g0.z = (unsigned)(ga & 0xffffffffull);               // global_addr[31:0]
      g0.w = (unsigned)((ga >> 32) & 0x01ffffffull)        // global_addr[56:32]
             | (2u << 30);                                 // type = 2 (image)
      const unsigned td0 = (unsigned)p.lda;
      v8i g1 = {};
      g1[0] = (int)((1u << 16)      // data_size = 2 bytes
                  | (1u << 20)      // pad_enable
                  | (4u << 22));    // pad_interval = 32 DWORDs (one row)
                                    // pad_amount[31:25] = 0 -> 1 DWORD
      g1[1] = (int)((td0 & 0xffffu) << 16);                // tensor_dim0[15:0]
      g1[2] = (int)((td0 >> 16) & 0xffffu);                // tensor_dim0[31:16]
      g1[3] = (int)(16u | (64u << 16));                    // tensor_dim1=1M; tile_dim0=64
      g1[4] = (int)128;                                    // tile_dim1=128 rows
      g1[5] = (int)td0;                                    // tensor_dim0_stride[31:0]
      g1[6] = 0;                                           // stride hi bits
      g1[7] = 0;
      v4i z4 = {};
#if __clang_major__ >= 23
      v8i z8 = {};
      __builtin_amdgcn_tensor_load_to_lds(g0, g1, z4, z4, z8, 0);
#else
      __builtin_amdgcn_tensor_load_to_lds(g0, g1, z4, z4, 0);
#endif
    }
#else
    for (int i = tid; i < 512; i += NT){
      const int r = i >> 2, ch = (i & 3) * 16;
      const unsigned short* src = A + (blockRow + r) * p.lda + k0 + ch;
      const uint4 d0 = *(const uint4*)(src);
      const uint4 d1 = *(const uint4*)(src + 8);
      unsigned int* dst = (unsigned int*)&As[r * LDSW + ch];
      dst[0] = d0.x; dst[1] = d0.y; dst[2] = d0.z; dst[3] = d0.w;
      dst[4] = d1.x; dst[5] = d1.y; dst[6] = d1.z; dst[7] = d1.w;
    }
#endif
    // ---- stage B tile (BN x 64 in Bs[n][k] layout) ----
    if (BT){ // B stored (N,K): direct copy
      for (int i = tid; i < BN * 8; i += NT){
        const int r = i >> 3, c = (i & 7) * 8;
        const unsigned short* src = B + (blockCol + r) * p.ldb + k0 + c;
        const uint4 d = *(const uint4*)src;
        unsigned int* dst = (unsigned int*)&Bs[r * LDSW + c];
        dst[0] = d.x; dst[1] = d.y; dst[2] = d.z; dst[3] = d.w;
      }
    } else { // B stored (K,N): coalesced read, transpose into Bs
      constexpr int PER = BN / 8;
      for (int i = tid; i < 64 * PER; i += NT){
        const int kk = i / PER, n0 = (i % PER) * 8;
        const unsigned short* src = B + (long)(k0 + kk) * p.ldb + blockCol + n0;
        const uint4 d = *(const uint4*)src;
        union { uint4 q; unsigned short s[8]; } tt; tt.q = d;
        #pragma unroll
        for (int j = 0; j < 8; ++j) Bs[(n0 + j) * LDSW + kk] = tt.s[j];
      }
    }
#if defined(HAVE_TDM)
    if (tid < 32) __builtin_amdgcn_s_wait_tensorcnt(0);
#endif
    __syncthreads();

    #pragma unroll
    for (int kc = 0; kc < 2; ++kc){
      FragB af[4], bfv[2];
      #pragma unroll
      for (int t2 = 0; t2 < 4; ++t2)
        frag_load(af[t2].u, &As[(waveM * 64 + t2 * 16 + rr) * LDSW + kc * 32], half);
      #pragma unroll
      for (int t2 = 0; t2 < 2; ++t2)
        frag_load(bfv[t2].u, &Bs[(waveN * 32 + t2 * 16 + rr) * LDSW + kc * 32], half);
      #pragma unroll
      for (int mt = 0; mt < 4; ++mt)
        #pragma unroll
        for (int nt = 0; nt < 2; ++nt)
          acc[mt][nt] = __builtin_amdgcn_wmma_f32_16x16x32_bf16(
              false, af[mt].v, false, bfv[nt].v, (short)0, acc[mt][nt], false, false);
    }
    __syncthreads();
  }

  // C layout: VGPR r -> M = r + half*8 ; N = lane&15
  #pragma unroll
  for (int mt = 0; mt < 4; ++mt)
    #pragma unroll
    for (int nt = 0; nt < 2; ++nt)
      #pragma unroll
      for (int r = 0; r < 8; ++r){
        const long gm = blockRow + waveM * 64 + mt * 16 + r + half * 8;
        const long gn = blockCol + waveN * 32 + nt * 16 + rr;
        const float val = acc[mt][nt][r] * p.alpha;
        if constexpr (EPI == 0){
          p.C[(long)blockIdx.z * p.sC + gm * p.ldc + gn] = val;
        } else if constexpr (EPI == 1){
          const int  seg = (int)(gn >> 9);
          const long o   = gm * CDIM + (gn & 511);
          if (seg == 0)      p.qout[o] = f2bf(val * 0.125f);   // q * DH^-0.5
          else if (seg == 1) p.kout[o] = f2bf(val);
          else { p.vout[o] = val; p.vbfout[o] = f2bf(val); }
        } else {
          const long hr = gm - p.rowoff;
          if (hr >= 0) p.Hres[hr * p.ldh + gn] += val + p.bias[gn];
        }
      }
}

// ---------------------------------------------------------------------------
// fp32 WMMA GEMM (16x16 tile per wave, K-step 4) — Newton-Schulz stays fp32.
// ---------------------------------------------------------------------------
struct GemmF {
  const float* A; long lda, sA;
  const float* B; long ldb, sB;
  float* C; long ldc, sC;
  int K; float alpha;
};

__global__ __launch_bounds__(32) void gemm_f32_k(GemmF p){
  const int lane = threadIdx.x;
  const int half = lane >> 4, rr = lane & 15;
  const long m0 = (long)blockIdx.y * 16;
  const long n0 = (long)blockIdx.x * 16;
  const float* A = p.A + (long)blockIdx.z * p.sA;
  const float* B = p.B + (long)blockIdx.z * p.sB;
  float*       C = p.C + (long)blockIdx.z * p.sC;
  v8f acc = {};
#if defined(HAVE_WMMA_F32X4)
  for (int k = 0; k < p.K; k += 4){
    v2f av = *(const v2f*)(A + (m0 + rr) * p.lda + k + half * 2);
    v2f bv;
    bv[0] = B[(long)(k + half * 2 + 0) * p.ldb + n0 + rr];
    bv[1] = B[(long)(k + half * 2 + 1) * p.ldb + n0 + rr];
    acc = __builtin_amdgcn_wmma_f32_16x16x4_f32(
        false, av, false, bv, (short)0, acc, false, false);
  }
#else
  for (int k = 0; k < p.K; ++k){
    const float bvv = B[(long)k * p.ldb + n0 + rr];
    #pragma unroll
    for (int r = 0; r < 8; ++r)
      acc[r] += A[(m0 + r + half * 8) * p.lda + k] * bvv;
  }
#endif
  #pragma unroll
  for (int r = 0; r < 8; ++r)
    C[(m0 + r + half * 8) * p.ldc + n0 + rr] = acc[r] * p.alpha;
}

// ---------------------------------------------------------------------------
// Elementwise / reduction helper kernels
// ---------------------------------------------------------------------------
__global__ void build_h_k(const float* x, const float* cls, float* H){
  long idx = (long)blockIdx.x * 256 + threadIdx.x;
  if (idx >= (long)NTOK * CDIM) return;
  H[idx] = (idx < CDIM) ? cls[idx] : x[idx - CDIM];
}

__global__ void cvt_f2bf_k(const float* X, unsigned short* Y, long n){
  long idx = (long)blockIdx.x * 256 + threadIdx.x;
  if (idx < n) Y[idx] = f2bf(X[idx]);
}

// LayerNorm row (512) -> bf16 with NPAD front zero padding.
__global__ __launch_bounds__(256) void ln_pad_k(const float* H, const float* w,
                                                const float* b, unsigned short* out){
  const int orow = blockIdx.x, tid = threadIdx.x;
  unsigned short* dst = out + (long)orow * CDIM;
  const int irow = orow - PADR;
  if (irow < 0){ dst[tid] = 0; dst[tid + 256] = 0; return; }
  const float* x = H + (long)irow * CDIM;
  const float v0 = x[tid], v1 = x[tid + 256];
  __shared__ float red[256];
  red[tid] = v0 + v1; __syncthreads();
  for (int s = 128; s > 0; s >>= 1){ if (tid < s) red[tid] += red[tid + s]; __syncthreads(); }
  const float mu = red[0] * (1.0f / CDIM); __syncthreads();
  const float d0 = v0 - mu, d1 = v1 - mu;
  red[tid] = d0 * d0 + d1 * d1; __syncthreads();
  for (int s = 128; s > 0; s >>= 1){ if (tid < s) red[tid] += red[tid + s]; __syncthreads(); }
  const float inv = rsqrtf(red[0] * (1.0f / CDIM) + 1e-5f);
  dst[tid]       = f2bf(d0 * inv * w[tid]       + b[tid]);
  dst[tid + 256] = f2bf(d1 * inv * w[tid + 256] + b[tid + 256]);
}

// LayerNorm row (512) -> f32, 1:1 rows (final norm).
__global__ __launch_bounds__(256) void ln_full_k(const float* H, const float* w,
                                                 const float* b, float* out){
  const int row = blockIdx.x, tid = threadIdx.x;
  const float* x = H + (long)row * CDIM;
  float* dst = out + (long)row * CDIM;
  const float v0 = x[tid], v1 = x[tid + 256];
  __shared__ float red[256];
  red[tid] = v0 + v1; __syncthreads();
  for (int s = 128; s > 0; s >>= 1){ if (tid < s) red[tid] += red[tid + s]; __syncthreads(); }
  const float mu = red[0] * (1.0f / CDIM); __syncthreads();
  const float d0 = v0 - mu, d1 = v1 - mu;
  red[tid] = d0 * d0 + d1 * d1; __syncthreads();
  for (int s = 128; s > 0; s >>= 1){ if (tid < s) red[tid] += red[tid + s]; __syncthreads(); }
  const float inv = rsqrtf(red[0] * (1.0f / CDIM) + 1e-5f);
  dst[tid]       = d0 * inv * w[tid]       + b[tid];
  dst[tid + 256] = d1 * inv * w[tid + 256] + b[tid + 256];
}

// Row softmax (in place, f32) + optional bf16 copy for WMMA operand use.
__global__ __launch_bounds__(256) void softmax_k(float* X, unsigned short* Xb, int cols){
  const long row = blockIdx.x;
  float* x = X + row * (long)cols;
  const int tid = threadIdx.x;
  __shared__ float red[256];
  float m = -3.4e38f;
  for (int c = tid; c < cols; c += 256) m = fmaxf(m, x[c]);
  red[tid] = m; __syncthreads();
  for (int s = 128; s > 0; s >>= 1){ if (tid < s) red[tid] = fmaxf(red[tid], red[tid + s]); __syncthreads(); }
  const float mx = red[0]; __syncthreads();
  float sum = 0.f;
  for (int c = tid; c < cols; c += 256) sum += __expf(x[c] - mx);
  red[tid] = sum; __syncthreads();
  for (int s = 128; s > 0; s >>= 1){ if (tid < s) red[tid] += red[tid + s]; __syncthreads(); }
  const float inv = 1.0f / red[0];
  for (int c = tid; c < cols; c += 256){
    const float v = __expf(x[c] - mx) * inv;
    x[c] = v;
    if (Xb) Xb[row * (long)cols + c] = f2bf(v);
  }
}

// Landmark means over LFOLD=40 consecutive rows (q already scaled).
__global__ void landmark_k(const unsigned short* Q, const unsigned short* K,
                           unsigned short* QL, unsigned short* KL){
  const int m = blockIdx.x, h = blockIdx.y, d = threadIdx.x;
  float sq = 0.f, sk = 0.f;
  const long base = (long)m * LFOLD * CDIM + h * DHEAD + d;
  for (int i = 0; i < LFOLD; ++i){
    sq += bf2f(Q[base + (long)i * CDIM]);
    sk += bf2f(K[base + (long)i * CDIM]);
  }
  const long o = ((long)h * NLM + m) * DHEAD + d;
  QL[o] = f2bf(sq * (1.0f / LFOLD));
  KL[o] = f2bf(sk * (1.0f / LFOLD));
}

// z0 = x^T / (max(row-abs-sum) * max(col-abs-sum)), per head.
__global__ __launch_bounds__(256) void zinit_k(const float* A2, float* Z){
  const int h = blockIdx.x, t = threadIdx.x;
  const float* a = A2 + (long)h * NLM * NLM;
  float* z = Z + (long)h * NLM * NLM;
  float rs = 0.f, cs = 0.f;
  for (int j = 0; j < NLM; ++j){
    rs += fabsf(a[t * NLM + j]);
    cs += fabsf(a[j * NLM + t]);
  }
  __shared__ float r1[256], r2[256];
  r1[t] = rs; r2[t] = cs; __syncthreads();
  for (int s = 128; s > 0; s >>= 1){
    if (t < s){ r1[t] = fmaxf(r1[t], r1[t + s]); r2[t] = fmaxf(r2[t], r2[t + s]); }
    __syncthreads();
  }
  const float scale = 1.0f / (r1[0] * r2[0]);
  for (int i = 0; i < NLM; ++i) z[t * NLM + i] = a[i * NLM + t] * scale;
}

// T = c*I - X  (batched 8 x 256 x 256)
__global__ void axpyI_k(const float* X, float* T, float c){
  const long idx = (long)blockIdx.x * 256 + threadIdx.x;
  const int ij = (int)(idx & 65535);
  T[idx] = ((ij >> 8) == (ij & 255) ? c : 0.f) - X[idx];
}

// OUTH += depthwise 33-tap conv of V along tokens; emit bf16 copy.
__global__ void convres_k(float* OUTH, unsigned short* OUTHB,
                          const float* V, const float* rw){
  const long idx = (long)blockIdx.x * 256 + threadIdx.x;
  const int n = (int)(idx >> 9), c = (int)(idx & 511), h = c >> 6;
  float acc = OUTH[idx];
  #pragma unroll
  for (int k = 0; k < 33; ++k){
    const int nn = n + k - 16;
    if (nn >= 0 && nn < NPAD) acc += V[(long)nn * CDIM + c] * rw[h * 33 + k];
  }
  OUTH[idx] = acc;
  OUTHB[idx] = f2bf(acc);
}

// PPEG: H[1+p] = F[p] + dw7 + b7 + dw5 + b5 + dw3 + b3 on 100x100 grid.
__global__ void ppeg_k(const float* F,
                       const float* w7, const float* b7,
                       const float* w5, const float* b5,
                       const float* w3, const float* b3,
                       float* H){
  const long idx = (long)blockIdx.x * 256 + threadIdx.x;
  const int p = (int)(idx >> 9), c = (int)(idx & 511);
  const int y = p / 100, x = p % 100;
  float acc = F[idx] + b7[c] + b5[c] + b3[c];
  const float* W7 = w7 + c * 49;
  const float* W5 = w5 + c * 25;
  const float* W3 = w3 + c * 9;
  for (int dy = -3; dy <= 3; ++dy){
    const int yy = y + dy; if (yy < 0 || yy >= 100) continue;
    for (int dx = -3; dx <= 3; ++dx){
      const int xx = x + dx; if (xx < 0 || xx >= 100) continue;
      acc += F[((yy * 100 + xx) << 9) + c] * W7[(dy + 3) * 7 + (dx + 3)];
    }
  }
  for (int dy = -2; dy <= 2; ++dy){
    const int yy = y + dy; if (yy < 0 || yy >= 100) continue;
    for (int dx = -2; dx <= 2; ++dx){
      const int xx = x + dx; if (xx < 0 || xx >= 100) continue;
      acc += F[((yy * 100 + xx) << 9) + c] * W5[(dy + 2) * 5 + (dx + 2)];
    }
  }
  for (int dy = -1; dy <= 1; ++dy){
    const int yy = y + dy; if (yy < 0 || yy >= 100) continue;
    for (int dx = -1; dx <= 1; ++dx){
      const int xx = x + dx; if (xx < 0 || xx >= 100) continue;
      acc += F[((yy * 100 + xx) << 9) + c] * W3[(dy + 1) * 3 + (dx + 1)];
    }
  }
  H[((long)(p + 1) << 9) + c] = acc;
}

// Final heads: row 0 -> bag logits; rows 1..10000 -> patch logits + softmax.
__global__ __launch_bounds__(256) void heads_k(const float* LN,
                                               const float* fcw, const float* fcb,
                                               const float* pcw, const float* pcb,
                                               float* out){
  const int r = blockIdx.x, t = threadIdx.x;
  const float* x = LN + (long)r * CDIM;
  const float* w  = (r == 0) ? fcw : pcw;
  const float* bb = (r == 0) ? fcb : pcb;
  float s0 = x[t] * w[t * 2]     + x[t + 256] * w[(t + 256) * 2];
  float s1 = x[t] * w[t * 2 + 1] + x[t + 256] * w[(t + 256) * 2 + 1];
  __shared__ float r0[256], r1[256];
  r0[t] = s0; r1[t] = s1; __syncthreads();
  for (int s = 128; s > 0; s >>= 1){
    if (t < s){ r0[t] += r0[t + s]; r1[t] += r1[t + s]; }
    __syncthreads();
  }
  if (t == 0){
    const float l0 = r0[0] + bb[0], l1 = r1[0] + bb[1];
    if (r == 0){ out[0] = l0; out[1] = l1; }
    else {
      const long o = 2 + (long)(r - 1) * 2;
      out[o] = l0; out[o + 1] = l1;
      const float m = fmaxf(l0, l1);
      const float e0 = __expf(l0 - m), e1 = __expf(l1 - m);
      const float inv = 1.0f / (e0 + e1);
      out[2 + 2L * NP + (long)(r - 1) * 2]     = e0 * inv;
      out[2 + 2L * NP + (long)(r - 1) * 2 + 1] = e1 * inv;
    }
  }
}

// ---------------------------------------------------------------------------
extern "C" void kernel_launch(void* const* d_in, const int* in_sizes, int n_in,
                              void* d_out, int out_size, void* d_ws, size_t ws_size,
                              hipStream_t stream) {
  (void)in_sizes; (void)n_in; (void)out_size; (void)ws_size;
  const float* x      = (const float*)d_in[0];
  const float* clstok = (const float*)d_in[1];
  const float* ln1w = (const float*)d_in[2];
  const float* ln1b = (const float*)d_in[3];
  const float* qkv1w = (const float*)d_in[4];
  const float* out1w = (const float*)d_in[5];
  const float* out1b = (const float*)d_in[6];
  const float* res1w = (const float*)d_in[7];
  const float* p7w = (const float*)d_in[8];
  const float* p7b = (const float*)d_in[9];
  const float* p5w = (const float*)d_in[10];
  const float* p5b = (const float*)d_in[11];
  const float* p3w = (const float*)d_in[12];
  const float* p3b = (const float*)d_in[13];
  const float* ln2w = (const float*)d_in[14];
  const float* ln2b = (const float*)d_in[15];
  const float* qkv2w = (const float*)d_in[16];
  const float* out2w = (const float*)d_in[17];
  const float* out2b = (const float*)d_in[18];
  const float* res2w = (const float*)d_in[19];
  const float* normw = (const float*)d_in[20];
  const float* normb = (const float*)d_in[21];
  const float* fc2w = (const float*)d_in[22];
  const float* fc2b = (const float*)d_in[23];
  const float* pcw = (const float*)d_in[24];
  const float* pcb = (const float*)d_in[25];

  char* ws = (char*)d_ws;
  size_t off = 0;
  auto alloc = [&](size_t bytes) -> void* {
    void* p = ws + off;
    off += (bytes + 255) & ~(size_t)255;
    return p;
  };

  float*          H      = (float*)alloc((size_t)NTOK * CDIM * 4);
  float*          FBUF   = (float*)alloc((size_t)NP * CDIM * 4);
  unsigned short* XPBF   = (unsigned short*)alloc((size_t)NPAD * CDIM * 2);
  unsigned short* WQKVBF = (unsigned short*)alloc((size_t)CDIM * 3 * CDIM * 2);
  unsigned short* OWBF   = (unsigned short*)alloc((size_t)CDIM * CDIM * 2);
  unsigned short* QBF    = (unsigned short*)alloc((size_t)NPAD * CDIM * 2);
  unsigned short* KBF    = (unsigned short*)alloc((size_t)NPAD * CDIM * 2);
  unsigned short* VBF    = (unsigned short*)alloc((size_t)NPAD * CDIM * 2);
  float*          Vf     = (float*)alloc((size_t)NPAD * CDIM * 4);
  unsigned short* QLBF   = (unsigned short*)alloc((size_t)NHEAD * NLM * DHEAD * 2);
  unsigned short* KLBF   = (unsigned short*)alloc((size_t)NHEAD * NLM * DHEAD * 2);
  float*          BIGF   = (float*)alloc((size_t)NHEAD * NPAD * NLM * 4);
  unsigned short* BIGBF  = (unsigned short*)alloc((size_t)NHEAD * NPAD * NLM * 2);
  float*          A2     = (float*)alloc((size_t)NHEAD * NLM * NLM * 4);
  float*          Zb     = (float*)alloc((size_t)NHEAD * NLM * NLM * 4);
  float*          Z2b    = (float*)alloc((size_t)NHEAD * NLM * NLM * 4);
  float*          XZ     = (float*)alloc((size_t)NHEAD * NLM * NLM * 4);
  float*          CB     = (float*)alloc((size_t)NHEAD * NLM * NLM * 4);
  float*          TB     = (float*)alloc((size_t)NHEAD * NLM * NLM * 4);
  float*          A3V    = (float*)alloc((size_t)NHEAD * NLM * DHEAD * 4);
  float*          W2     = (float*)alloc((size_t)NHEAD * NLM * DHEAD * 4);
  unsigned short* W2BF   = (unsigned short*)alloc((size_t)NHEAD * NLM * DHEAD * 2);
  float*          OUTH   = (float*)alloc((size_t)NPAD * CDIM * 4);
  unsigned short* OUTHBF = (unsigned short*)alloc((size_t)NPAD * CDIM * 2);
  float*          LNF    = OUTH;  // reuse: dead by the time final LN runs

  auto gemmF = [&](float* C, long ldc, long sC,
                   const float* A, long lda, long sA,
                   const float* B, long ldb, long sB,
                   int M, int N, int K, float alpha){
    GemmF g{};
    g.A = A; g.lda = lda; g.sA = sA;
    g.B = B; g.ldb = ldb; g.sB = sB;
    g.C = C; g.ldc = ldc; g.sC = sC;
    g.K = K; g.alpha = alpha;
    gemm_f32_k<<<dim3(N / 16, M / 16, NHEAD), 32, 0, stream>>>(g);
  };

  auto attn_block = [&](const float* lnw, const float* lnb, const float* qkvw,
                        const float* outw, const float* outb, const float* resw){
    // weight conversion
    cvt_f2bf_k<<<(CDIM * 3 * CDIM + 255) / 256, 256, 0, stream>>>(qkvw, WQKVBF, (long)CDIM * 3 * CDIM);
    cvt_f2bf_k<<<(CDIM * CDIM + 255) / 256, 256, 0, stream>>>(outw, OWBF, (long)CDIM * CDIM);
    // layernorm + front pad + bf16
    ln_pad_k<<<NPAD, 256, 0, stream>>>(H, lnw, lnb, XPBF);
    // QKV projection with split epilogue (10240 x 1536 x 512)
    {
      GemmP p{};
      p.A = XPBF; p.lda = CDIM; p.sA = 0;
      p.B = WQKVBF; p.ldb = 3 * CDIM; p.sB = 0;
      p.M = NPAD; p.N = 3 * CDIM; p.K = CDIM; p.alpha = 1.f;
      p.vout = Vf; p.qout = QBF; p.kout = KBF; p.vbfout = VBF;
      gemm_bf16_k<1, false, 4><<<dim3(3 * CDIM / 128, NPAD / 128, 1), 256, 0, stream>>>(p);
    }
    landmark_k<<<dim3(NLM, NHEAD), DHEAD, 0, stream>>>(QBF, KBF, QLBF, KLBF);
    // attn3 logits = q_l @ k^T  (256 x 10240 per head)
    {
      GemmP p{};
      p.A = QLBF; p.lda = DHEAD; p.sA = (long)NLM * DHEAD;
      p.B = KBF;  p.ldb = CDIM;  p.sB = DHEAD;            // NT: rows = token n
      p.C = BIGF; p.ldc = NPAD;  p.sC = (long)NLM * NPAD;
      p.M = NLM; p.N = NPAD; p.K = DHEAD; p.alpha = 1.f;
      gemm_bf16_k<0, true, 4><<<dim3(NPAD / 128, NLM / 128, NHEAD), 256, 0, stream>>>(p);
    }
    softmax_k<<<NHEAD * NLM, 256, 0, stream>>>(BIGF, BIGBF, NPAD);
    // a3v = attn3 @ v  (256 x 64 per head, K = 10240)
    {
      GemmP p{};
      p.A = BIGBF; p.lda = NPAD; p.sA = (long)NLM * NPAD;
      p.B = VBF;   p.ldb = CDIM; p.sB = DHEAD;            // NN: rows = token k
      p.C = A3V;   p.ldc = DHEAD; p.sC = (long)NLM * DHEAD;
      p.M = NLM; p.N = DHEAD; p.K = NPAD; p.alpha = 1.f;
      gemm_bf16_k<0, false, 2><<<dim3(1, NLM / 128, NHEAD), 128, 0, stream>>>(p);
    }
    // attn2 logits = q_l @ k_l^T (256 x 256)
    {
      GemmP p{};
      p.A = QLBF; p.lda = DHEAD; p.sA = (long)NLM * DHEAD;
      p.B = KLBF; p.ldb = DHEAD; p.sB = (long)NLM * DHEAD;
      p.C = A2;   p.ldc = NLM;   p.sC = (long)NLM * NLM;
      p.M = NLM; p.N = NLM; p.K = DHEAD; p.alpha = 1.f;
      gemm_bf16_k<0, true, 4><<<dim3(NLM / 128, NLM / 128, NHEAD), 256, 0, stream>>>(p);
    }
    softmax_k<<<NHEAD * NLM, 256, 0, stream>>>(A2, nullptr, NLM);
    // Newton-Schulz pseudo-inverse, fp32 WMMA
    zinit_k<<<NHEAD, 256, 0, stream>>>(A2, Zb);
    float* Zc = Zb; float* Zn = Z2b;
    const long SQ = (long)NLM * NLM;
    const int axg = (int)(NHEAD * SQ / 256);
    for (int it = 0; it < 6; ++it){
      gemmF(XZ, NLM, SQ, A2, NLM, SQ, Zc, NLM, SQ, NLM, NLM, NLM, 1.f);
      axpyI_k<<<axg, 256, 0, stream>>>(XZ, TB, 7.f);
      gemmF(CB, NLM, SQ, XZ, NLM, SQ, TB, NLM, SQ, NLM, NLM, NLM, 1.f);
      axpyI_k<<<axg, 256, 0, stream>>>(CB, TB, 15.f);
      gemmF(CB, NLM, SQ, XZ, NLM, SQ, TB, NLM, SQ, NLM, NLM, NLM, 1.f);
      axpyI_k<<<axg, 256, 0, stream>>>(CB, TB, 13.f);
      gemmF(Zn, NLM, SQ, Zc, NLM, SQ, TB, NLM, SQ, NLM, NLM, NLM, 0.25f);
      float* t = Zc; Zc = Zn; Zn = t;
    }
    // w2 = pinv @ (attn3 @ v)
    gemmF(W2, DHEAD, (long)NLM * DHEAD, Zc, NLM, SQ, A3V, DHEAD, (long)NLM * DHEAD,
          NLM, DHEAD, NLM, 1.f);
    cvt_f2bf_k<<<(NHEAD * NLM * DHEAD + 255) / 256, 256, 0, stream>>>(
        W2, W2BF, (long)NHEAD * NLM * DHEAD);
    // attn1 logits = q @ k_l^T  (10240 x 256 per head)
    {
      GemmP p{};
      p.A = QBF;  p.lda = CDIM;  p.sA = DHEAD;
      p.B = KLBF; p.ldb = DHEAD; p.sB = (long)NLM * DHEAD;
      p.C = BIGF; p.ldc = NLM;   p.sC = (long)NPAD * NLM;
      p.M = NPAD; p.N = NLM; p.K = DHEAD; p.alpha = 1.f;
      gemm_bf16_k<0, true, 4><<<dim3(NLM / 128, NPAD / 128, NHEAD), 256, 0, stream>>>(p);
    }
    softmax_k<<<NHEAD * NPAD, 256, 0, stream>>>(BIGF, BIGBF, NLM);
    // head outputs = attn1 @ w2 -> OUTH[(n, 64h+d)]
    {
      GemmP p{};
      p.A = BIGBF; p.lda = NLM;   p.sA = (long)NPAD * NLM;
      p.B = W2BF;  p.ldb = DHEAD; p.sB = (long)NLM * DHEAD;
      p.C = OUTH;  p.ldc = CDIM;  p.sC = DHEAD;
      p.M = NPAD; p.N = DHEAD; p.K = NLM; p.alpha = 1.f;
      gemm_bf16_k<0, false, 2><<<dim3(1, NPAD / 128, NHEAD), 128, 0, stream>>>(p);
    }
    // + depthwise residual conv of v, emit bf16 operand
    convres_k<<<NPAD * CDIM / 256, 256, 0, stream>>>(OUTH, OUTHBF, Vf, resw);
    // out projection + bias + residual into H (drop the PADR front rows)
    {
      GemmP p{};
      p.A = OUTHBF; p.lda = CDIM; p.sA = 0;
      p.B = OWBF;   p.ldb = CDIM; p.sB = 0;
      p.M = NPAD; p.N = CDIM; p.K = CDIM; p.alpha = 1.f;
      p.bias = outb; p.Hres = H; p.rowoff = PADR; p.ldh = CDIM;
      gemm_bf16_k<2, false, 4><<<dim3(CDIM / 128, NPAD / 128, 1), 256, 0, stream>>>(p);
    }
  };

  // ---- pipeline ----
  build_h_k<<<((long)NTOK * CDIM + 255) / 256, 256, 0, stream>>>(x, clstok, H);
  attn_block(ln1w, ln1b, qkv1w, out1w, out1b, res1w);
  hipMemcpyAsync(FBUF, H + CDIM, (size_t)NP * CDIM * 4, hipMemcpyDeviceToDevice, stream);
  ppeg_k<<<NP * CDIM / 256, 256, 0, stream>>>(FBUF, p7w, p7b, p5w, p5b, p3w, p3b, H);
  attn_block(ln2w, ln2b, qkv2w, out2w, out2b, res2w);
  ln_full_k<<<NTOK, 256, 0, stream>>>(H, normw, normb, LNF);
  heads_k<<<NTOK, 256, 0, stream>>>(LNF, fc2w, fc2b, pcw, pcb, (float*)d_out);
}
